// EIGNN_scale_w_iter_52733608461014
// MI455X (gfx1250) — matrix operand order repaired and checked by reading, not compiled, not run on hardware
//
#include <hip/hip_runtime.h>

// ---------------------------------------------------------------------------
// EIGNN forward on MI455X (gfx1250, wave32, WMMA + async LDS staging).
//   G  = F^T F / (||F^T F||_F + 1e-12), loop-invariant
//   Z <- gamma * G @ (Z @ S) + X        (fixed 20 applications from Z = X)
// Z@S is L2-resident (S = 151MB < 192MB L2) and compute-limited in f32 ->
// f16 WMMA (16x16x32) with f32 accumulation. A-tiles stage via
// GLOBAL_LOAD_ASYNC_TO_LDS_B128 (ASYNCcnt), B-tiles stage manually because
// they need an f32->f16 K-transposed layout for b128 fragment reads.
// ---------------------------------------------------------------------------

typedef __attribute__((ext_vector_type(16))) _Float16 v16h;
typedef __attribute__((ext_vector_type(8)))  float    v8f;

#define MDIM 512
#define NDIM 6144
#define GAMMA_F 0.8f
#define EPS_F 1e-12f
#define LDSP 40            // padded f16 LDS row stride in halves (32 + 8)
#define APF  36            // padded f32 LDS row stride in floats (32 + 4)

#if __has_builtin(__builtin_amdgcn_global_load_async_to_lds_b128)
#define HAVE_ASYNC 1
#else
#define HAVE_ASYNC 0
#endif

union HF16 { v16h v; unsigned u[8]; };

__device__ __forceinline__ unsigned pack2h(_Float16 a, _Float16 b) {
  union { _Float16 h[2]; unsigned u; } x;
  x.h[0] = a; x.h[1] = b; return x.u;
}

#if HAVE_ASYNC
// exact param types per hipcc diagnostic: (int4 AS1*, int4 AS3*, imm, imm)
typedef int v4i_ __attribute__((vector_size(16)));
typedef __attribute__((address_space(1))) v4i_ g_v4i;
typedef __attribute__((address_space(3))) v4i_ l_v4i;
__device__ __forceinline__ void async_b128(const void* g, void* l) {
  __builtin_amdgcn_global_load_async_to_lds_b128((g_v4i*)g, (l_v4i*)l, 0, 0);
}
__device__ __forceinline__ void wait_async_all() {
#if __has_builtin(__builtin_amdgcn_s_wait_asynccnt)
  __builtin_amdgcn_s_wait_asynccnt(0);
#else
  asm volatile("s_wait_asynccnt 0x0" ::: "memory");
#endif
}
#endif

// ---- FF = F^T @ F  (512 x 512, f32 exact) ---------------------------------
__global__ __launch_bounds__(256) void ff_kernel(const float* __restrict__ F,
                                                 float* __restrict__ FF) {
  __shared__ float fa[16][17];
  __shared__ float fb[16][17];
  const int tx = threadIdx.x, ty = threadIdx.y;
  const int i0 = blockIdx.y * 16, j0 = blockIdx.x * 16;
  float acc = 0.f;
  for (int k0 = 0; k0 < MDIM; k0 += 16) {
    fa[ty][tx] = F[(k0 + ty) * MDIM + i0 + tx];
    fb[ty][tx] = F[(k0 + ty) * MDIM + j0 + tx];
    __syncthreads();
#pragma unroll
    for (int k = 0; k < 16; ++k) acc += fa[k][ty] * fb[k][tx];
    __syncthreads();
  }
  FF[(i0 + ty) * MDIM + j0 + tx] = acc;
}

// ---- sum of squares of FF -> *accum (atomicAdd) ---------------------------
__global__ __launch_bounds__(256) void normsq_kernel(const float* __restrict__ FF,
                                                     float* __restrict__ accum) {
  __shared__ float red[256];
  float s = 0.f;
  for (int i = blockIdx.x * 256 + threadIdx.x; i < MDIM * MDIM;
       i += 256 * gridDim.x) {
    float v = FF[i];
    s += v * v;
  }
  red[threadIdx.x] = s;
  __syncthreads();
  for (int off = 128; off > 0; off >>= 1) {
    if (threadIdx.x < off) red[threadIdx.x] += red[threadIdx.x + off];
    __syncthreads();
  }
  if (threadIdx.x == 0) atomicAdd(accum, red[0]);
}

// ---- Gs = f16( gamma * FF / (sqrt(sumsq)+eps) ) ---------------------------
__global__ __launch_bounds__(256) void scale_kernel(const float* __restrict__ FF,
                                                    const float* __restrict__ sumsq,
                                                    _Float16* __restrict__ Gs) {
  const float sc = GAMMA_F / (sqrtf(*sumsq) + EPS_F);
  const int i = blockIdx.x * 256 + threadIdx.x;
  Gs[i] = (_Float16)(FF[i] * sc);
}

// ---- W = Z @ S   (M=512, N=6144, K=6144) ----------------------------------
// block: 256 thr (8 wave32), tile 64(M) x 128(N), BK=32; wave = 16x64 strip.
// A tile staged f32 via async-to-LDS; converted to f16 at fragment build.
__global__ __launch_bounds__(256) void zs_gemm_kernel(const float* __restrict__ Z,
                                                      const float* __restrict__ S,
                                                      float* __restrict__ W) {
  __shared__ __align__(16) float     aF[64 * APF];   // [row m][k]  (f32)
  __shared__ __align__(16) _Float16  bT[128 * LDSP]; // [col n][k]  (f16, K-transposed)
  const int tid  = threadIdx.x;
  const int lane = tid & 31, wave = tid >> 5;
  const int wm = wave & 3, wn = wave >> 2;
  const int m0b = blockIdx.y * 64;
  const int n0b = blockIdx.x * 128;
  const int g = lane >> 4;
  const int l15 = lane & 15;

  v8f acc[4] = {};

  for (int k0 = 0; k0 < NDIM; k0 += 32) {
    if (k0 + 32 < NDIM) {  // warm next S tile (global_prefetch_b8)
      __builtin_prefetch(&S[(k0 + 32 + (tid >> 3)) * NDIM + n0b + (tid & 7) * 16], 0, 1);
    }
    // stage A: Z[m0b..+64)[k0..+32) f32 (async direct to LDS when available)
#pragma unroll
    for (int j = 0; j < 2; ++j) {
      const int s = j * 256 + tid;          // 512 float4 slots
      const int r = s >> 3;
      const int c = (s & 7) * 4;
#if HAVE_ASYNC
      async_b128(&Z[(m0b + r) * NDIM + k0 + c], &aF[r * APF + c]);
#else
      *(float4*)&aF[r * APF + c] =
          *(const float4*)&Z[(m0b + r) * NDIM + k0 + c];
#endif
    }
    // stage B: S[k0..+32)[n0b..+128) -> f16 transposed [n][k]
#pragma unroll
    for (int j = 0; j < 4; ++j) {
      const int s = j * 256 + tid;          // 1024 float4 slots
      const int r = s >> 5;
      const int c = (s & 31) * 4;
      const float4 v = *(const float4*)&S[(k0 + r) * NDIM + n0b + c];
      bT[(c + 0) * LDSP + r] = (_Float16)v.x;
      bT[(c + 1) * LDSP + r] = (_Float16)v.y;
      bT[(c + 2) * LDSP + r] = (_Float16)v.z;
      bT[(c + 3) * LDSP + r] = (_Float16)v.w;
    }
#if HAVE_ASYNC
    wait_async_all();
#endif
    __syncthreads();

    // A fragment: row (lane&15); K per ISA 16-bit A layout, cvt f32->f16 here
    const float* ar = &aF[(wm * 16 + l15) * APF];
    const float4 q0 = *(const float4*)(ar + 8 * g);
    const float4 q1 = *(const float4*)(ar + 8 * g + 4);
    const float4 q2 = *(const float4*)(ar + 16 + 8 * g);
    const float4 q3 = *(const float4*)(ar + 16 + 8 * g + 4);
    HF16 af;
    af.u[0] = pack2h((_Float16)q0.x, (_Float16)q0.y);
    af.u[1] = pack2h((_Float16)q0.z, (_Float16)q0.w);
    af.u[2] = pack2h((_Float16)q1.x, (_Float16)q1.y);
    af.u[3] = pack2h((_Float16)q1.z, (_Float16)q1.w);
    af.u[4] = pack2h((_Float16)q2.x, (_Float16)q2.y);
    af.u[5] = pack2h((_Float16)q2.z, (_Float16)q2.w);
    af.u[6] = pack2h((_Float16)q3.x, (_Float16)q3.y);
    af.u[7] = pack2h((_Float16)q3.z, (_Float16)q3.w);

#pragma unroll
    for (int tn = 0; tn < 4; ++tn) {
      HF16 bf;
      const int bcol = wn * 64 + tn * 16 + l15;
#pragma unroll
      for (int p = 0; p < 8; ++p) {
        const int kb = 2 * p + 8 * g + ((p >= 4) ? 8 : 0);
        bf.u[p] = *(const unsigned*)&bT[bcol * LDSP + kb];
      }
      acc[tn] = __builtin_amdgcn_wmma_f32_16x16x32_f16(
          false, af.v, false, bf.v, (short)0, acc[tn], false, false);
    }
    __syncthreads();
  }

#pragma unroll
  for (int tn = 0; tn < 4; ++tn)
#pragma unroll
    for (int v = 0; v < 8; ++v) {
      const int m = m0b + wm * 16 + v + 8 * g;
      const int n = n0b + wn * 64 + tn * 16 + l15;
      W[m * NDIM + n] = acc[tn][v];
    }
}

// ---- dst = Gs(f16) @ W + X   (M=512, N=6144, K=512) -----------------------
__global__ __launch_bounds__(256) void gw_gemm_kernel(const _Float16* __restrict__ Gs,
                                                      const float* __restrict__ Wm,
                                                      const float* __restrict__ X,
                                                      float* __restrict__ dst) {
  __shared__ __align__(16) _Float16 aT[64 * LDSP];
  __shared__ __align__(16) _Float16 bT[128 * LDSP];
  const int tid  = threadIdx.x;
  const int lane = tid & 31, wave = tid >> 5;
  const int wm = wave & 3, wn = wave >> 2;
  const int m0b = blockIdx.y * 64;
  const int n0b = blockIdx.x * 128;
  const int g = lane >> 4;
  const int l15 = lane & 15;

  v8f acc[4] = {};

  for (int k0 = 0; k0 < MDIM; k0 += 32) {
    // stage A: Gs already f16 row-major; one b128 (8 halves) per thread
    {
      const int r = tid >> 2;
      const int c = (tid & 3) * 8;
#if HAVE_ASYNC
      async_b128(&Gs[(m0b + r) * MDIM + k0 + c], &aT[r * LDSP + c]);
#else
      *(uint4*)&aT[r * LDSP + c] =
          *(const uint4*)&Gs[(m0b + r) * MDIM + k0 + c];
#endif
    }
    // stage B: W f32 -> f16 transposed [n][k]
#pragma unroll
    for (int j = 0; j < 4; ++j) {
      const int s = j * 256 + tid;
      const int r = s >> 5;
      const int c = (s & 31) * 4;
      const float4 v = *(const float4*)&Wm[(k0 + r) * NDIM + n0b + c];
      bT[(c + 0) * LDSP + r] = (_Float16)v.x;
      bT[(c + 1) * LDSP + r] = (_Float16)v.y;
      bT[(c + 2) * LDSP + r] = (_Float16)v.z;
      bT[(c + 3) * LDSP + r] = (_Float16)v.w;
    }
#if HAVE_ASYNC
    wait_async_all();
#endif
    __syncthreads();

    HF16 af;
    const int arow = wm * 16 + l15;
#pragma unroll
    for (int p = 0; p < 8; ++p) {
      const int kb = 2 * p + 8 * g + ((p >= 4) ? 8 : 0);
      af.u[p] = *(const unsigned*)&aT[arow * LDSP + kb];
    }
#pragma unroll
    for (int tn = 0; tn < 4; ++tn) {
      HF16 bf;
      const int bcol = wn * 64 + tn * 16 + l15;
#pragma unroll
      for (int p = 0; p < 8; ++p) {
        const int kb = 2 * p + 8 * g + ((p >= 4) ? 8 : 0);
        bf.u[p] = *(const unsigned*)&bT[bcol * LDSP + kb];
      }
      acc[tn] = __builtin_amdgcn_wmma_f32_16x16x32_f16(
          false, af.v, false, bf.v, (short)0, acc[tn], false, false);
    }
    __syncthreads();
  }

#pragma unroll
  for (int tn = 0; tn < 4; ++tn)
#pragma unroll
    for (int v = 0; v < 8; ++v) {
      const int m = m0b + wm * 16 + v + 8 * g;
      const int n = n0b + wn * 64 + tn * 16 + l15;
      dst[m * NDIM + n] = acc[tn][v] + X[m * NDIM + n];
    }
}

// ---------------------------------------------------------------------------
extern "C" void kernel_launch(void* const* d_in, const int* in_sizes, int n_in,
                              void* d_out, int out_size, void* d_ws, size_t ws_size,
                              hipStream_t stream) {
  const float* X = (const float*)d_in[0];   // [512, 6144]
  const float* F = (const float*)d_in[1];   // [512, 512]
  const float* S = (const float*)d_in[2];   // [6144, 6144]
  float* out = (float*)d_out;               // [512, 6144]

  char* ws = (char*)d_ws;
  // workspace layout (~26 MB total)
  float*    FF    = (float*)(ws + 0);                        // 1 MB
  _Float16* Gs    = (_Float16*)(ws + (1u << 20));            // 0.5 MB
  float*    nrm   = (float*)(ws + (1u << 20) + (512u << 10));// 4 B
  float*    Z     = (float*)(ws + (1u << 20) + (512u << 10) + 256u);
  float*    W     = Z + (size_t)MDIM * NDIM;

  const size_t zs_bytes = (size_t)MDIM * NDIM * sizeof(float);

  // G = gamma * F^T F / (||F^T F||_F + eps), folded into f16 Gs
  (void)hipMemsetAsync(nrm, 0, sizeof(float), stream);
  ff_kernel<<<dim3(32, 32), dim3(16, 16), 0, stream>>>(F, FF);
  normsq_kernel<<<64, 256, 0, stream>>>(FF, nrm);
  scale_kernel<<<(MDIM * MDIM) / 256, 256, 0, stream>>>(FF, nrm, Gs);

  // Z1 = inner(0) = X
  (void)hipMemcpyAsync(Z, X, zs_bytes, hipMemcpyDeviceToDevice, stream);

  // 19 solver steps (z2..z20; reference hits MAX_ITER) + 1 output application
  const dim3 grid(NDIM / 128, MDIM / 64);
  for (int it = 0; it < 20; ++it) {
    zs_gemm_kernel<<<grid, 256, 0, stream>>>(Z, S, W);
    float* dst = (it == 19) ? out : Z;
    gw_gemm_kernel<<<grid, 256, 0, stream>>>(Gs, W, X, dst);
  }
}
